// VRPActionNet_66924180407124
// MI455X (gfx1250) — matrix operand order repaired and checked by reading, not compiled, not run on hardware
//
#include <hip/hip_runtime.h>
#include <hip/hip_bf16.h>

typedef __attribute__((ext_vector_type(16))) _Float16 v16h;
typedef __attribute__((ext_vector_type(8)))  _Float16 v8h;
typedef __attribute__((ext_vector_type(8)))  float    v8f;

#define DEVINL __device__ __forceinline__

// Padded LDS row stride for 128-wide activation tiles: 136 f16 = 272 B.
// Keeps 16B alignment for ds_load_b128 while avoiding the 64-bank wrap
// (128 f16 = 256 B = exactly 64 banks -> 16-way conflicts).
#define LDA128 136

// ---------------------------------------------------------------------------
// WMMA fragment helpers, per CDNA5 ISA 7.12.2 wave32 layouts.
// ---------------------------------------------------------------------------

// A-matrix 16x32 f16 (MxK): lanes 0-15 M=lane (K-halves 0/16), lanes 16-31 M=lane-16 (K-halves 8/24)
// Two contiguous 16B runs per lane -> 2x ds_load_b128.
DEVINL v16h load_A_frag_lds(const _Float16* a, int lda, int k0) {
    int lane = threadIdx.x & 31;
    int m  = lane & 15;
    int kh = (lane & 16) ? 8 : 0;
    const _Float16* row = a + m * lda + k0 + kh;
    v8h lo = *(const v8h*)(row);
    v8h hi = *(const v8h*)(row + 16);
    v16h out;
#pragma unroll
    for (int i = 0; i < 8; ++i) { out[i] = lo[i]; out[8 + i] = hi[i]; }
    return out;
}

// B fragments are pre-swizzled in the workspace: for tile (kt, nt), lane l's
// 16 values are contiguous -> one 32B load (2x global_load_b128) per lane.
// packed index = (((kt*8 + nt)*32 + lane)*16 + e)
DEVINL v8f gemm_tile_packed(const _Float16* a_lds, int lda,
                            const _Float16* wp, int Kdim, int nt, v8f c) {
    int lane = threadIdx.x & 31;
    for (int kt = 0; kt < (Kdim >> 5); ++kt) {
        v16h a = load_A_frag_lds(a_lds, lda, kt * 32);
        v16h b = *(const v16h*)(wp + ((((size_t)kt * 8 + nt) * 32 + lane) << 4));
        c = __builtin_amdgcn_wmma_f32_16x16x32_f16(false, a, false, b, (short)0, c, false, false);
    }
    return c;
}

// C/D 16x16 f32: VGPR r holds M = r (lanes 0-15) or M = 8 + r (lanes 16-31), N = lane&15
DEVINL v8f bias_init(const float* bias, int n0) {
    int lane = threadIdx.x & 31;
    float bv = bias[n0 + (lane & 15)];
    v8f c;
#pragma unroll
    for (int r = 0; r < 8; ++r) c[r] = bv;
    return c;
}

DEVINL void store_D_lds_f16(_Float16* out, int ldo, v8f d, int n0, bool relu) {
    int lane = threadIdx.x & 31;
    int n  = n0 + (lane & 15);
    int mh = (lane & 16) ? 8 : 0;
#pragma unroll
    for (int r = 0; r < 8; ++r) {
        float v = d[r];
        if (relu) v = fmaxf(v, 0.0f);
        out[(mh + r) * ldo + n] = (_Float16)v;
    }
}

DEVINL void store_D_global_rows_f16(_Float16* base, size_t row0, v8f d, int n0) {
    int lane = threadIdx.x & 31;
    int n  = n0 + (lane & 15);
    int mh = (lane & 16) ? 8 : 0;
#pragma unroll
    for (int r = 0; r < 8; ++r)
        base[(row0 + mh + r) * 128 + n] = (_Float16)d[r];
}

DEVINL void cvt8_f32_to_f16(const float* __restrict__ sp, _Float16* __restrict__ dst) {
    float4 f0 = *(const float4*)(sp);
    float4 f1 = *(const float4*)(sp + 4);
    v8h o;
    o[0] = (_Float16)f0.x; o[1] = (_Float16)f0.y; o[2] = (_Float16)f0.z; o[3] = (_Float16)f0.w;
    o[4] = (_Float16)f1.x; o[5] = (_Float16)f1.y; o[6] = (_Float16)f1.z; o[7] = (_Float16)f1.w;
    *(v8h*)dst = o;
}

// ---------------------------------------------------------------------------
// Kernels
// ---------------------------------------------------------------------------

// Convert f32 weight [K][128] into f16 WMMA-fragment order (see gemm_tile_packed).
__global__ void k_pack_w(const float* __restrict__ src, _Float16* __restrict__ dst, int K) {
    int o = blockIdx.x * blockDim.x + threadIdx.x;
    if (o >= K * 128) return;
    int e    = o & 15;
    int lane = (o >> 4) & 31;
    int t    = o >> 9;            // kt*8 + nt
    int nt   = t & 7;
    int kt   = t >> 3;
    int kh   = (lane & 16) ? 8 : 0;
    int k    = kt * 32 + kh + (e & 7) + ((e >> 3) << 4);
    int n    = nt * 16 + (lane & 15);
    dst[o] = (_Float16)src[k * 128 + n];
}

// First-move logits: gather 16 rows (node or edge embeddings), 3x [128x128] ReLU layers, [128->1] head.
__global__ void __launch_bounds__(256)
k_first(const float* __restrict__ x_emb, const float* __restrict__ e_emb,
        const int* __restrict__ nodes, const int* __restrict__ edges,
        const _Float16* __restrict__ fmW, const float* __restrict__ fm_b,
        const float* __restrict__ fm_Wo, const float* __restrict__ fm_bo,
        float* __restrict__ out) {
    __shared__ _Float16 bufA[16 * LDA128];
    __shared__ _Float16 bufB[16 * LDA128];
    __shared__ const float* srcp[16];
    __shared__ int outoff[16];
    __shared__ float red[256];

    int tid = threadIdx.x;
    int gr0 = blockIdx.x * 16;

    if (tid < 16) {
        int gr = gr0 + tid;
        const float* p;
        int oo;
        if (gr < 16 * 256) {                       // node rows
            int b = gr >> 8, i = gr & 255;
            int node = nodes[b * 256 + i];
            p  = x_emb + ((size_t)b * 200 + node) * 128;
            oo = b * 3584 + i;
        } else {                                   // edge rows
            int ge = gr - 16 * 256;
            int b = ge >> 8, m = ge & 255;
            int ii = edges[(b * 256 + m) * 2 + 0];
            int jj = edges[(b * 256 + m) * 2 + 1];
            p  = e_emb + (((size_t)b * 200 + ii) * 200 + jj) * 128;
            oo = b * 3584 + 256 + m;
        }
        __builtin_prefetch(p, 0, 0);               // global_prefetch_b8
        srcp[tid]   = p;
        outoff[tid] = oo;
    }
    __syncthreads();

    // vectorized gather+convert: 16 rows x 16 chunks of 8 f32
    for (int idx = tid; idx < 16 * 16; idx += 256) {
        int r = idx >> 4, ch = idx & 15;
        cvt8_f32_to_f16(srcp[r] + ch * 8, bufA + r * LDA128 + ch * 8);
    }
    __syncthreads();

    _Float16* cur = bufA;
    _Float16* nxt = bufB;
    int wave = tid >> 5;
    for (int layer = 0; layer < 3; ++layer) {
        v8f c = bias_init(fm_b + layer * 128, wave * 16);
        c = gemm_tile_packed(cur, LDA128, fmW + layer * 16384, 128, wave, c);
        store_D_lds_f16(nxt, LDA128, c, wave * 16, true);
        __syncthreads();
        _Float16* t = cur; cur = nxt; nxt = t;
    }

    // [128 -> 1] head: 16 lanes per row, 8 elems each, LDS reduce
    {
        int r = tid >> 4, t = tid & 15;
        const _Float16* row = cur + r * LDA128 + t * 8;
        const float* wo = fm_Wo + t * 8;
        float s = 0.0f;
#pragma unroll
        for (int q = 0; q < 8; ++q) s += (float)row[q] * wo[q];
        red[tid] = s;
    }
    __syncthreads();
    if (tid < 16) {
        float acc = fm_bo[0];
#pragma unroll
        for (int q = 0; q < 16; ++q) acc += red[tid * 16 + q];
        out[outoff[tid]] = acc;
    }
}

// Cost embedding: h1 = relu(cost * cW1 + cb1); cemb = h1 @ cW2 + cb2  -> f16 ws
__global__ void __launch_bounds__(256)
k_cemb(const float* __restrict__ rcost, const float* __restrict__ ccost,
       const float* __restrict__ tcost,
       const float* __restrict__ cW1, const float* __restrict__ cb1,
       const _Float16* __restrict__ cW2, const float* __restrict__ cb2,
       _Float16* __restrict__ cemb) {
    __shared__ _Float16 h1[16 * LDA128];
    int tid = threadIdx.x;
    int gr0 = blockIdx.x * 16;

    for (int idx = tid; idx < 16 * 128; idx += 256) {
        int r = idx >> 7, h = idx & 127;
        int gr = gr0 + r;
        int type = gr >> 14;                       // 16384 rows per move type
        int bm = gr & 16383;
        const float* cp = (type == 0) ? rcost : (type == 1) ? ccost : tcost;
        float v = cp[bm] * cW1[h] + cb1[h];
        h1[r * LDA128 + h] = (_Float16)fmaxf(v, 0.0f);
    }
    __syncthreads();

    int wave = tid >> 5;
    v8f c = bias_init(cb2, wave * 16);
    c = gemm_tile_packed(h1, LDA128, cW2, 128, wave, c);
    store_D_global_rows_f16(cemb, (size_t)gr0, c, wave * 16);
}

// Move embedding: gather k edge rows + cemb row -> [16 x (k+1)*128] tile, two GEMMs -> all_emb (f16)
__global__ void __launch_bounds__(256)
k_moveemb(const float* __restrict__ e_emb,
          const int* __restrict__ ridx, const int* __restrict__ cidx,
          const int* __restrict__ tidx,
          const _Float16* __restrict__ cemb,
          const _Float16* __restrict__ rW1, const float* __restrict__ rb1,
          const _Float16* __restrict__ rW2, const float* __restrict__ rb2,
          const _Float16* __restrict__ xW1, const float* __restrict__ xb1,
          const _Float16* __restrict__ xW2, const float* __restrict__ xb2,
          _Float16* __restrict__ allemb) {
    __shared__ _Float16 abuf[16 * (896 + 8)];      // padded stride K1+8
    __shared__ _Float16 hbuf[16 * LDA128];
    __shared__ const void* segp[16][7];

    int tid = threadIdx.x;
    int gr0 = blockIdx.x * 16;
    int type = gr0 >> 14;                          // block never straddles types
    int k    = (type == 0) ? 6 : 4;
    int segs = k + 1;
    int K1   = segs * 128;                         // 896 or 640
    int ldaA = K1 + 8;
    const int*      idxarr = (type == 0) ? ridx : (type == 1) ? cidx : tidx;
    const _Float16* W1 = (type == 0) ? rW1 : xW1;
    const float*    b1 = (type == 0) ? rb1 : xb1;
    const _Float16* W2 = (type == 0) ? rW2 : xW2;
    const float*    b2 = (type == 0) ? rb2 : xb2;

    for (int p = tid; p < 16 * segs; p += 256) {
        int r = p / segs, s = p % segs;
        int gr = gr0 + r;
        int bm = gr & 16383;
        if (s < k) {
            int ii = idxarr[((size_t)bm * k + s) * 2 + 0];
            int jj = idxarr[((size_t)bm * k + s) * 2 + 1];
            int b = bm >> 10;
            const float* ep = e_emb + (((size_t)b * 200 + ii) * 200 + jj) * 128;
            __builtin_prefetch(ep, 0, 0);
            segp[r][s] = ep;
        } else {
            segp[r][s] = cemb + (size_t)gr * 128;
        }
    }
    __syncthreads();

    // vectorized gather: 8-element chunks; f32 segs convert, f16 seg copies
    for (int idx = tid; idx < 16 * segs * 16; idx += 256) {
        int r   = idx / (segs * 16);
        int rem = idx % (segs * 16);
        int s   = rem >> 4;
        int ch  = rem & 15;
        _Float16* dst = abuf + r * ldaA + s * 128 + ch * 8;
        if (s < k) {
            cvt8_f32_to_f16(((const float*)segp[r][s]) + ch * 8, dst);
        } else {
            *(v8h*)dst = *(const v8h*)(((const _Float16*)segp[r][s]) + ch * 8);
        }
    }
    __syncthreads();

    int wave = tid >> 5;
    v8f c1 = bias_init(b1, wave * 16);
    c1 = gemm_tile_packed(abuf, ldaA, W1, K1, wave, c1);
    store_D_lds_f16(hbuf, LDA128, c1, wave * 16, true);
    __syncthreads();

    v8f c2 = bias_init(b2, wave * 16);
    c2 = gemm_tile_packed(hbuf, LDA128, W2, 128, wave, c2);
    store_D_global_rows_f16(allemb, (size_t)gr0, c2, wave * 16);
}

// Second-move logits: an-MLP (3x 128x128 ReLU) + [128->1] head over all_emb rows.
__global__ void __launch_bounds__(256)
k_second(const _Float16* __restrict__ allemb,
         const _Float16* __restrict__ anW, const float* __restrict__ an_b,
         const float* __restrict__ an_Wo, const float* __restrict__ an_bo,
         float* __restrict__ out) {
    __shared__ _Float16 bufA[16 * LDA128];
    __shared__ _Float16 bufB[16 * LDA128];
    __shared__ float red[256];
    int tid = threadIdx.x;
    int gr0 = blockIdx.x * 16;

    // 16B-chunk copy into padded LDS
    for (int idx = tid; idx < 16 * 16; idx += 256) {
        int r = idx >> 4, ch = idx & 15;
        *(v8h*)(bufA + r * LDA128 + ch * 8) =
            *(const v8h*)(allemb + ((size_t)(gr0 + r)) * 128 + ch * 8);
    }
    __syncthreads();

    _Float16* cur = bufA;
    _Float16* nxt = bufB;
    int wave = tid >> 5;
    for (int layer = 0; layer < 3; ++layer) {
        v8f c = bias_init(an_b + layer * 128, wave * 16);
        c = gemm_tile_packed(cur, LDA128, anW + layer * 16384, 128, wave, c);
        store_D_lds_f16(nxt, LDA128, c, wave * 16, true);
        __syncthreads();
        _Float16* t = cur; cur = nxt; nxt = t;
    }

    {
        int r = tid >> 4, t = tid & 15;
        const _Float16* row = cur + r * LDA128 + t * 8;
        const float* wo = an_Wo + t * 8;
        float s = 0.0f;
#pragma unroll
        for (int q = 0; q < 8; ++q) s += (float)row[q] * wo[q];
        red[tid] = s;
    }
    __syncthreads();
    if (tid < 16) {
        int gr = gr0 + tid;
        int typ = gr >> 14;
        int bm = gr & 16383;
        int b = bm >> 10, m = bm & 1023;
        float acc = an_bo[0];
#pragma unroll
        for (int q = 0; q < 16; ++q) acc += red[tid * 16 + q];
        out[(size_t)b * 3584 + 512 + typ * 1024 + m] = acc;
    }
}

// ---------------------------------------------------------------------------
// Launch
// ---------------------------------------------------------------------------

extern "C" void kernel_launch(void* const* d_in, const int* in_sizes, int n_in,
                              void* d_out, int out_size, void* d_ws, size_t ws_size,
                              hipStream_t stream) {
    const float* x_emb  = (const float*)d_in[0];
    const float* e_emb  = (const float*)d_in[1];
    const int*   nodes  = (const int*)d_in[2];
    const int*   edges  = (const int*)d_in[3];
    const int*   ridx   = (const int*)d_in[4];
    const float* rcost  = (const float*)d_in[5];
    const int*   cidx   = (const int*)d_in[6];
    const float* ccost  = (const float*)d_in[7];
    const int*   tidx   = (const int*)d_in[8];
    const float* tcost  = (const float*)d_in[9];
    const float* fm_W   = (const float*)d_in[10];
    const float* fm_b   = (const float*)d_in[11];
    const float* fm_Wo  = (const float*)d_in[12];
    const float* fm_bo  = (const float*)d_in[13];
    const float* c_W1   = (const float*)d_in[14];
    const float* c_b1   = (const float*)d_in[15];
    const float* c_W2   = (const float*)d_in[16];
    const float* c_b2   = (const float*)d_in[17];
    const float* r_W1   = (const float*)d_in[18];
    const float* r_b1   = (const float*)d_in[19];
    const float* r_W2   = (const float*)d_in[20];
    const float* r_b2   = (const float*)d_in[21];
    const float* x_W1   = (const float*)d_in[22];
    const float* x_b1   = (const float*)d_in[23];
    const float* x_W2   = (const float*)d_in[24];
    const float* x_b2   = (const float*)d_in[25];
    const float* an_W   = (const float*)d_in[26];
    const float* an_b   = (const float*)d_in[27];
    const float* an_Wo  = (const float*)d_in[28];
    const float* an_bo  = (const float*)d_in[29];
    (void)in_sizes; (void)n_in; (void)out_size; (void)ws_size;

    float* out = (float*)d_out;

    // Workspace layout (f16 elements, every region a multiple of 16 elems -> 32B aligned)
    _Float16* ws = (_Float16*)d_ws;
    size_t off = 0;
    _Float16* fmW16  = ws + off; off += 3 * 128 * 128;   // packed fragments
    _Float16* cW2_16 = ws + off; off += 128 * 128;
    _Float16* rW1_16 = ws + off; off += 896 * 128;
    _Float16* rW2_16 = ws + off; off += 128 * 128;
    _Float16* xW1_16 = ws + off; off += 640 * 128;
    _Float16* xW2_16 = ws + off; off += 128 * 128;
    _Float16* anW16  = ws + off; off += 3 * 128 * 128;
    _Float16* cemb   = ws + off; off += (size_t)49152 * 128;   // cost embeddings
    _Float16* allemb = ws + off; off += (size_t)49152 * 128;   // move embeddings

    // Stage 0: weights -> f16 in WMMA fragment order
    auto pack = [&](const float* s, _Float16* d, int K) {
        int n = K * 128;
        k_pack_w<<<(n + 255) / 256, 256, 0, stream>>>(s, d, K);
    };
    pack(fm_W, fmW16, 3 * 128);     // three stacked 128x128 layers pack contiguously
    pack(c_W2, cW2_16, 128);
    pack(r_W1, rW1_16, 896);
    pack(r_W2, rW2_16, 128);
    pack(x_W1, xW1_16, 640);
    pack(x_W2, xW2_16, 128);
    pack(an_W, anW16, 3 * 128);

    // Stage 1: first-move logits (8192 rows, 16 per block)
    k_first<<<512, 256, 0, stream>>>(x_emb, e_emb, nodes, edges,
                                     fmW16, fm_b, fm_Wo, fm_bo, out);

    // Stage 2: cost embeddings (49152 rows)
    k_cemb<<<3072, 256, 0, stream>>>(rcost, ccost, tcost, c_W1, c_b1,
                                     cW2_16, c_b2, cemb);

    // Stage 3: move embeddings (49152 rows)
    k_moveemb<<<3072, 256, 0, stream>>>(e_emb, ridx, cidx, tidx, cemb,
                                        rW1_16, r_b1, rW2_16, r_b2,
                                        xW1_16, x_b1, xW2_16, x_b2, allemb);

    // Stage 4: second-move logits (49152 rows)
    k_second<<<3072, 256, 0, stream>>>(allemb, anW16, an_b, an_Wo, an_bo, out);
}
